// C2QAttention_17746804867735
// MI455X (gfx1250) — matrix operand order repaired and checked by reading, not compiled, not run on hardware
//
#include <hip/hip_runtime.h>
#include <math.h>

typedef __attribute__((ext_vector_type(16))) _Float16 v16h;
typedef __attribute__((ext_vector_type(8)))  _Float16 v8h;
typedef __attribute__((ext_vector_type(8)))  float    v8f;

#define B_N  32
#define C_N  2048
#define Q_N  512
#define H_N  1024

#define CT 64                 // C rows per workgroup
#define HT 128                // H cols per workgroup
#define KT 32                 // K step (f16 WMMA K)
#define KSTEPS (Q_N / KT)     // 16

#define A_ROW   40            // halves per A-chunk row (32 + 8 pad -> 80B stride)
#define A_SLOT  (CT * A_ROW)  // halves per A slot
#define E_ROW   40            // halves per E row   (32 + 8 pad -> 80B stride)
#define E_SLOT  (HT * E_ROW)  // halves per E slot

__global__ __launch_bounds__(256)
void c2q_attn_wmma(const float* __restrict__ sim,
                   const float* __restrict__ qe,
                   float* __restrict__ out)
{
    __shared__ __align__(16) _Float16 Ach[2 * A_SLOT]; // exp(S - max) tiles, f16
    __shared__ __align__(16) _Float16 Et [2 * E_SLOT]; // qencode tiles, f16, [h][k]
    __shared__ float Sc[CT];                           // per-row sum of exp

    const int tid = threadIdx.x;
    const int b   = blockIdx.z;
    const int c0  = blockIdx.y * CT;
    const int h0  = blockIdx.x * HT;

    // ---- softmax stats mapping: 4 lanes per C-row, 128 Q elems each ----
    const int r    = tid >> 2;   // 0..63
    const int part = tid & 3;    // 0..3

    // Phase 1: per-row max over Q (single pass, float4 loads)
    const float4* rowp =
        (const float4*)(sim + (size_t)(b * C_N + c0 + r) * Q_N) + part * 32;
    float m = -INFINITY;
    #pragma unroll 8
    for (int i = 0; i < 32; ++i) {
        float4 v = rowp[i];
        m = fmaxf(m, fmaxf(fmaxf(v.x, v.y), fmaxf(v.z, v.w)));
    }
    m = fmaxf(m, __shfl_xor(m, 1));
    m = fmaxf(m, __shfl_xor(m, 2));
    if (part == 0) Sc[r] = 0.f;

    // ---- E-staging mapping: one H column of 16 K values per thread-half ----
    const int he = tid & 127;        // H column within tile
    const int ke = (tid >> 7) * 16;  // K sub-block (0 or 16)

    float ssum = 0.f;

    auto stage = [&](int ks, int slot) {
        // A chunk: 64 rows x 32 K of exp(S - rowmax), stored f16
        {
            const float4* gp =
                (const float4*)(sim + (size_t)(b * C_N + c0 + r) * Q_N + ks * KT) + part * 2;
            float4 u = gp[0];
            float4 v = gp[1];
            float e0 = __expf(u.x - m), e1 = __expf(u.y - m);
            float e2 = __expf(u.z - m), e3 = __expf(u.w - m);
            float e4 = __expf(v.x - m), e5 = __expf(v.y - m);
            float e6 = __expf(v.z - m), e7 = __expf(v.w - m);
            ssum += ((e0 + e1) + (e2 + e3)) + ((e4 + e5) + (e6 + e7));
            v8h h = { (_Float16)e0, (_Float16)e1, (_Float16)e2, (_Float16)e3,
                      (_Float16)e4, (_Float16)e5, (_Float16)e6, (_Float16)e7 };
            *(v8h*)(Ach + slot * A_SLOT + r * A_ROW + part * 8) = h;
        }
        // E chunk: 32 K x 128 H, stored K-contiguous per column: Et[h][k]
        {
            const float* gp = qe + (size_t)(b * Q_N + ks * KT + ke) * H_N + h0 + he;
            _Float16* lp = Et + slot * E_SLOT + he * E_ROW + ke;
            v8h lo, hi;
            #pragma unroll
            for (int i = 0; i < 8; ++i) lo[i] = (_Float16)gp[i * H_N];
            #pragma unroll
            for (int i = 0; i < 8; ++i) hi[i] = (_Float16)gp[(8 + i) * H_N];
            *(v8h*)lp = lo;
            *(v8h*)(lp + 8) = hi;
        }
    };

    // ---- GEMM wave mapping: wave w -> 16 C-rows x 64 H-cols (4 WMMA tiles) ----
    const int w    = tid >> 5;
    const int lane = tid & 31;
    const int l16  = lane & 15;
    const int sel  = lane >> 4;        // 0: lanes 0-15, 1: lanes 16-31
    const int mrow = (w & 3) * 16;     // C-row tile base
    const int ncb  = (w >> 2) * 64;    // H-col base (4 x 16 tiles)

    v8f acc0 = {}, acc1 = {}, acc2 = {}, acc3 = {};

    __syncthreads();          // Sc initialized
    stage(0, 0);
    __syncthreads();          // slot 0 ready

    #pragma unroll 1
    for (int ks = 0; ks < KSTEPS; ++ks) {
        const int cur = ks & 1;
        if (ks + 1 < KSTEPS) stage(ks + 1, cur ^ 1);

        // A fragment (contiguous K: lanes<16 K0-15, lanes>=16 K16-31)
        const _Float16* ap = Ach + cur * A_SLOT + (mrow + l16) * A_ROW + sel * 16;
        v8h a0 = *(const v8h*)ap;          // K base + 0..7
        v8h a1 = *(const v8h*)(ap + 8);    // K base + 8..15
        v16h a = __builtin_shufflevector(a0, a1,
                 0,1,2,3,4,5,6,7,8,9,10,11,12,13,14,15);

        // B fragments (contiguous K, matching A: lanes<16 K0-15, lanes>=16 K16-31; N = lane&15)
        const _Float16* eb = Et + cur * E_SLOT + sel * 16;
        {
            const _Float16* bp = eb + (ncb + 0 + l16) * E_ROW;
            v8h b0 = *(const v8h*)bp;
            v8h b1 = *(const v8h*)(bp + 8);
            v16h bf = __builtin_shufflevector(b0, b1,
                      0,1,2,3,4,5,6,7,8,9,10,11,12,13,14,15);
            acc0 = __builtin_amdgcn_wmma_f32_16x16x32_f16(
                       false, a, false, bf, (short)0, acc0, false, false);
        }
        {
            const _Float16* bp = eb + (ncb + 16 + l16) * E_ROW;
            v8h b0 = *(const v8h*)bp;
            v8h b1 = *(const v8h*)(bp + 8);
            v16h bf = __builtin_shufflevector(b0, b1,
                      0,1,2,3,4,5,6,7,8,9,10,11,12,13,14,15);
            acc1 = __builtin_amdgcn_wmma_f32_16x16x32_f16(
                       false, a, false, bf, (short)0, acc1, false, false);
        }
        {
            const _Float16* bp = eb + (ncb + 32 + l16) * E_ROW;
            v8h b0 = *(const v8h*)bp;
            v8h b1 = *(const v8h*)(bp + 8);
            v16h bf = __builtin_shufflevector(b0, b1,
                      0,1,2,3,4,5,6,7,8,9,10,11,12,13,14,15);
            acc2 = __builtin_amdgcn_wmma_f32_16x16x32_f16(
                       false, a, false, bf, (short)0, acc2, false, false);
        }
        {
            const _Float16* bp = eb + (ncb + 48 + l16) * E_ROW;
            v8h b0 = *(const v8h*)bp;
            v8h b1 = *(const v8h*)(bp + 8);
            v16h bf = __builtin_shufflevector(b0, b1,
                      0,1,2,3,4,5,6,7,8,9,10,11,12,13,14,15);
            acc3 = __builtin_amdgcn_wmma_f32_16x16x32_f16(
                       false, a, false, bf, (short)0, acc3, false, false);
        }
        __syncthreads();
    }

    // Row sums (unnormalized exp) -> Sc, then normalize in epilogue
    atomicAdd(&Sc[r], ssum);
    __syncthreads();

    const int rbase = mrow + sel * 8;  // acc reg i -> C-row rbase+i
    #pragma unroll
    for (int i = 0; i < 8; ++i) {
        float s = 1.0f / Sc[rbase + i];
        float* orow = out + (size_t)(b * C_N + c0 + rbase + i) * H_N + h0 + ncb + l16;
        orow[0]  = acc0[i] * s;
        orow[16] = acc1[i] * s;
        orow[32] = acc2[i] * s;
        orow[48] = acc3[i] * s;
    }
}

extern "C" void kernel_launch(void* const* d_in, const int* in_sizes, int n_in,
                              void* d_out, int out_size, void* d_ws, size_t ws_size,
                              hipStream_t stream) {
    const float* sim = (const float*)d_in[0];  // [B, C, Q] f32
    const float* qe  = (const float*)d_in[1];  // [B, Q, H] f32
    float* out = (float*)d_out;                // [B, C, H] f32
    dim3 grid(H_N / HT, C_N / CT, B_N);        // (8, 32, 32)
    dim3 block(256);
    c2q_attn_wmma<<<grid, block, 0, stream>>>(sim, qe, out);
}
